// TransformerLayer_36558761623844
// MI455X (gfx1250) — matrix operand order repaired and checked by reading, not compiled, not run on hardware
//
#include <hip/hip_runtime.h>
#include <hip/hip_bf16.h>

// ---------------- problem constants (fixed shapes from the reference) ----------
#define Bb   4
#define Ss   2048
#define Dd   512
#define Hh   8
#define DHd  64
#define Mm   64
#define DI   2048
#define RR   (Bb * Ss)          // 8192 rows
#define NP   128                // padded value-dim (64 v + 1 ones + 63 zero)
#define FEAT 256                // m * 2P = 64*4

// ---------------- bf16 helpers (raw ushort storage) ----------------------------
typedef __attribute__((ext_vector_type(16))) __bf16 v16bf;
typedef __attribute__((ext_vector_type(8)))  float  v8f;

union FragBF { v16bf v; unsigned short u[16]; };

__device__ __forceinline__ unsigned short f2bf(float f) {
  unsigned u = __float_as_uint(f);
  u += 0x7FFFu + ((u >> 16) & 1u);      // round-to-nearest-even
  return (unsigned short)(u >> 16);
}
__device__ __forceinline__ float bf2f(unsigned short h) {
  return __uint_as_float(((unsigned)h) << 16);
}

// CDNA5 async copy: global memory -> LDS, 16 bytes per lane, tracked by ASYNCcnt.
// LDS destination address = low 32 bits of the generic pointer (aperture mapping).
__device__ __forceinline__ void async_copy16(const unsigned short* g, unsigned short* l) {
  unsigned loff = (unsigned)(unsigned long)l;
  asm volatile("global_load_async_to_lds_b128 %0, %1, off"
               :: "v"(loff), "v"(g) : "memory");
}
// ASYNCcnt decrements in issue order: waiting <=4 retires the previous tile's
// 4 loads while the newest tile's 4 loads remain in flight (double buffering).
__device__ __forceinline__ void async_wait0() {
  asm volatile("s_wait_asynccnt 0x0" ::: "memory");
}
__device__ __forceinline__ void async_wait4() {
  asm volatile("s_wait_asynccnt 0x4" ::: "memory");
}

// ---------------- generic WMMA GEMM: C = A(bf16) @ Bt(bf16)^T [+bias][relu|elu+1][+resid]
// A:  M x K row-major (lda = K-stride)
// Bt: N x K row-major (ldb = K-stride)  — i.e. B stored transposed
// C:  M x N (ldc). Batched via blockIdx.z.
// Block: 128 threads (4 waves). Block tile 64x64, wave tile 32x32 (2x2 WMMA frags).
// Double-buffered LDS staging driven by the ASYNCcnt pipeline.
#define BM 64
#define BN 64
#define BK 32

template<bool BIAS, bool RELU, bool ELUP1, bool RESID, bool OUTBF16>
__global__ __launch_bounds__(128) void gemm_wmma(
    const unsigned short* __restrict__ A, const unsigned short* __restrict__ Bt,
    const float* __restrict__ bias, const float* __restrict__ resid,
    void* __restrict__ Cv,
    int M, int N, int K, int lda, int ldb, int ldc,
    long aBatch, long bBatch, long cBatch)
{
  __shared__ unsigned short As[2][BM][BK];   // A tile, row-major (m, k)
  __shared__ unsigned short Bs[2][BN][BK];   // B tile, row-major (n, k)

  const int bz = blockIdx.z;
  A  += (long)bz * aBatch;
  Bt += (long)bz * bBatch;

  const int tile_n = blockIdx.x * BN;
  const int tile_m = blockIdx.y * BM;
  const int tid  = threadIdx.x;
  const int wave = tid >> 5;
  const int lane = tid & 31;
  const int wr = (wave >> 1) * 32;   // wave row offset within block tile
  const int wc = (wave & 1)  * 32;   // wave col offset
  const int lrow = lane & 15;
  const int lhi  = lane >> 4;        // 0 or 1

  v8f acc[2][2];
#pragma unroll
  for (int i = 0; i < 2; ++i)
#pragma unroll
    for (int j = 0; j < 2; ++j)
#pragma unroll
      for (int r = 0; r < 8; ++r) acc[i][j][r] = 0.0f;

  const int r = tid >> 1;            // 0..63: row of A tile / row of Bt tile
  const int c = (tid & 1) * 16;      // 0 or 16: k-chunk
  const unsigned short* arow = A  + (long)(tile_m + r) * lda + c;
  const unsigned short* brow = Bt + (long)(tile_n + r) * ldb + c;

  auto issue = [&](int k0, int buf) {
    async_copy16(arow + k0,     &As[buf][r][c]);
    async_copy16(arow + k0 + 8, &As[buf][r][c + 8]);
    async_copy16(brow + k0,     &Bs[buf][r][c]);
    async_copy16(brow + k0 + 8, &Bs[buf][r][c + 8]);
  };

  const int nt = K / BK;
  issue(0, 0);                        // prologue: tile 0 -> buffer 0

  for (int t = 0; t < nt; ++t) {
    const int buf = t & 1;
    if (t + 1 < nt) {
      issue((t + 1) * BK, buf ^ 1);   // overlap: tile t+1 -> other buffer
      async_wait4();                  // tile t's 4 loads retired; t+1 in flight
      if (t + 2 < nt) {               // warm L2 one stage ahead of the DMA
        __builtin_prefetch(arow + (t + 2) * BK, 0, 1);   // global_prefetch_b8
        __builtin_prefetch(brow + (t + 2) * BK, 0, 1);
      }
    } else {
      async_wait0();
    }
    __syncthreads();                  // publish buf across waves

    // ---- build fragments per ISA 16-bit layouts and issue 4 WMMAs
    FragBF af[2], bfm[2];
#pragma unroll
    for (int i = 0; i < 2; ++i) {
      const int ar = wr + i * 16 + lrow;
      const unsigned short* ap = &As[buf][ar][lhi * 8];
#pragma unroll
      for (int j = 0; j < 8; ++j) { af[i].u[j] = ap[j]; af[i].u[8 + j] = ap[16 + j]; }
      const int bcol = wc + i * 16 + lrow;
      const unsigned short* bp = &Bs[buf][bcol][lhi * 8];
#pragma unroll
      for (int j = 0; j < 8; ++j) { bfm[i].u[j] = bp[j]; bfm[i].u[8 + j] = bp[16 + j]; }
    }
#pragma unroll
    for (int i = 0; i < 2; ++i)
#pragma unroll
      for (int j = 0; j < 2; ++j)
        acc[i][j] = __builtin_amdgcn_wmma_f32_16x16x32_bf16(
            false, af[i].v, false, bfm[j].v, (short)0, acc[i][j], false, false);
    __syncthreads();                  // all reads of buf done before it is reused
  }

  // ---- epilogue
#pragma unroll
  for (int i = 0; i < 2; ++i) {
#pragma unroll
    for (int j = 0; j < 2; ++j) {
      const int col = tile_n + wc + j * 16 + lrow;
#pragma unroll
      for (int rr = 0; rr < 8; ++rr) {
        const int row = tile_m + wr + i * 16 + lhi * 8 + rr;
        float v = acc[i][j][rr];
        if (BIAS)  v += bias[col];
        if (RELU)  v = v > 0.0f ? v : 0.0f;
        if (ELUP1) v = v > 0.0f ? v + 1.0f : __expf(v);   // elu(x)+1
        if (RESID) v += resid[(long)row * ldc + col];
        const long cidx = (long)bz * cBatch + (long)row * ldc + col;
        if (OUTBF16) ((unsigned short*)Cv)[cidx] = f2bf(v);
        else         ((float*)Cv)[cidx] = v;
      }
    }
  }
}

// ---------------- elementwise kernels ------------------------------------------
// fp32 (Rr x Cc, row-major) -> bf16 transposed (Cc x Rr, row-major)
__global__ void cvtT_bf16_kernel(const float* __restrict__ in,
                                 unsigned short* __restrict__ out, int Rr, int Cc) {
  long i = (long)blockIdx.x * blockDim.x + threadIdx.x;
  if (i < (long)Rr * Cc) {
    const int rr = (int)(i / Cc);
    const int cc = (int)(i % Cc);
    out[(long)cc * Rr + rr] = f2bf(in[i]);
  }
}

// per-row mean subtraction; fp32 in -> bf16 centered out. Block = 256, one row/block.
__global__ void center_bf16_kernel(const float* __restrict__ in,
                                   unsigned short* __restrict__ out, int D) {
  __shared__ float red[256];
  const long row = blockIdx.x;
  const float* r = in + row * D;
  float s = 0.0f;
  for (int j = threadIdx.x; j < D; j += 256) s += r[j];
  red[threadIdx.x] = s;
  __syncthreads();
  for (int off = 128; off > 0; off >>= 1) {
    if ((int)threadIdx.x < off) red[threadIdx.x] += red[threadIdx.x + off];
    __syncthreads();
  }
  const float mean = red[0] / (float)D;
  for (int j = threadIdx.x; j < D; j += 256) out[row * D + j] = f2bf(r[j] - mean);
}

// Fq[(b,h)][s][m*4+p] and FkT[(b,h)][m*4+p][s] = phi * pfeat(s,h,p)
__global__ void buildF_kernel(const unsigned short* __restrict__ phiq,
                              const unsigned short* __restrict__ phik,
                              const float* __restrict__ pos_ft,
                              const float* __restrict__ slopes,
                              unsigned short* __restrict__ Fq,
                              unsigned short* __restrict__ FkT) {
  long idx = (long)blockIdx.x * blockDim.x + threadIdx.x;   // over B*H*S*64
  if (idx >= (long)Bb * Hh * Ss * Mm) return;
  const int m = idx & 63;
  long rest = idx >> 6;
  const int s = rest & (Ss - 1); rest >>= 11;
  const int h = rest & (Hh - 1);
  const int b = rest >> 3;

  const float slope = slopes[h];
  const float a0 = pos_ft[s * 2 + 0] * slope;
  const float a1 = pos_ft[s * 2 + 1] * slope;
  const float inv = 0.70710678118f;   // 1/sqrt(P), P=2
  float pf[4] = { __cosf(a0) * inv, __cosf(a1) * inv,
                  __sinf(a0) * inv, __sinf(a1) * inv };

  const long prow = ((long)(b * Ss + s)) * (Hh * Mm) + h * Mm + m;
  const float q = bf2f(phiq[prow]);
  const float k = bf2f(phik[prow]);

  const long bh = (long)b * Hh + h;
  const long fqbase = (bh * Ss + s) * FEAT + m * 4;
  const long fkbase = (bh * FEAT + m * 4) * Ss + s;
#pragma unroll
  for (int p = 0; p < 4; ++p) {
    Fq[fqbase + p]       = f2bf(q * pf[p]);
    FkT[fkbase + p * Ss] = f2bf(k * pf[p]);
  }
}

// V_ext transposed: VextT[(b,h)][d(0..127)][s] ; d<64: v, d==64: 1, else 0
__global__ void buildVT_kernel(const unsigned short* __restrict__ qkv,
                               unsigned short* __restrict__ VextT) {
  long idx = (long)blockIdx.x * blockDim.x + threadIdx.x;   // over B*H*128*S
  if (idx >= (long)Bb * Hh * NP * Ss) return;
  const int s = idx & (Ss - 1);
  long rest = idx >> 11;
  const int d  = rest & (NP - 1);
  const long bh = rest >> 7;
  const int h = (int)(bh & (Hh - 1));
  const int b = (int)(bh >> 3);
  unsigned short v;
  if (d < DHd)      v = qkv[((long)(b * Ss + s)) * (3 * Hh * DHd) + 2 * Hh * DHd + h * DHd + d];
  else if (d == 64) v = 0x3F80;   // 1.0 in bf16
  else              v = 0;
  VextT[idx] = v;
}

// attn[b,s,h*64+d] = num[(b,h)][s][d] / (num[(b,h)][s][64] + eps)   (bf16 out)
__global__ void attn_norm_kernel(const float* __restrict__ num,
                                 unsigned short* __restrict__ attn) {
  long idx = (long)blockIdx.x * blockDim.x + threadIdx.x;   // over B*S*H*64
  if (idx >= (long)Bb * Ss * Hh * DHd) return;
  const int d = idx & 63;
  long rest = idx >> 6;
  const int h = rest & (Hh - 1); rest >>= 3;
  const int s = rest & (Ss - 1);
  const int b = rest >> 11;
  const long base = (((long)b * Hh + h) * Ss + s) * NP;
  const float n = num[base + d];
  const float z = num[base + 64];
  attn[((long)(b * Ss + s)) * (Hh * DHd) + h * DHd + d] = f2bf(n / (z + 1e-6f));
}

// ---------------- launch -------------------------------------------------------
extern "C" void kernel_launch(void* const* d_in, const int* in_sizes, int n_in,
                              void* d_out, int out_size, void* d_ws, size_t ws_size,
                              hipStream_t stream) {
  const float* inp    = (const float*)d_in[0];
  const float* pos_ft = (const float*)d_in[1];
  const float* slopes = (const float*)d_in[2];
  const float* Wqkv   = (const float*)d_in[3];
  const float* bqkv   = (const float*)d_in[4];
  const float* Wmap   = (const float*)d_in[5];
  const float* Wo     = (const float*)d_in[6];
  const float* bo     = (const float*)d_in[7];
  const float* W1     = (const float*)d_in[8];
  const float* b1     = (const float*)d_in[9];
  const float* W2     = (const float*)d_in[10];
  const float* b2     = (const float*)d_in[11];
  float* out = (float*)d_out;

  char* ws = (char*)d_ws;
  size_t off = 0;
  auto take = [&](size_t bytes) -> char* {
    char* p = ws + off;
    off = (off + bytes + 255) & ~(size_t)255;
    return p;
  };

  unsigned short* wqkvT   = (unsigned short*)take((size_t)Dd * 3 * Hh * DHd * 2);  // 1536 x 512
  unsigned short* wmapT   = (unsigned short*)take((size_t)Hh * DHd * Mm * 2);      // per-head 64 x 64
  unsigned short* woT     = (unsigned short*)take((size_t)Hh * DHd * Dd * 2);      // 512 x 512
  unsigned short* w1T     = (unsigned short*)take((size_t)Dd * DI * 2);            // 2048 x 512
  unsigned short* w2T     = (unsigned short*)take((size_t)DI * Dd * 2);            // 512 x 2048
  unsigned short* ain_bf  = (unsigned short*)take((size_t)RR * Dd * 2);
  unsigned short* qkv_bf  = (unsigned short*)take((size_t)RR * 3 * Hh * DHd * 2);
  unsigned short* phiq_bf = (unsigned short*)take((size_t)RR * Hh * Mm * 2);
  unsigned short* phik_bf = (unsigned short*)take((size_t)RR * Hh * Mm * 2);
  unsigned short* Fq      = (unsigned short*)take((size_t)Bb * Hh * Ss * FEAT * 2);
  unsigned short* FkT     = (unsigned short*)take((size_t)Bb * Hh * FEAT * Ss * 2);
  unsigned short* VextT   = (unsigned short*)take((size_t)Bb * Hh * NP * Ss * 2);
  unsigned short* KVT     = (unsigned short*)take((size_t)Bb * Hh * NP * FEAT * 2);
  float*          numb    = (float*)take((size_t)Bb * Hh * Ss * NP * 4);
  unsigned short* attn_bf = (unsigned short*)take((size_t)RR * Hh * DHd * 2);
  float*          xbuf    = (float*)take((size_t)RR * Dd * 4);
  unsigned short* ffin_bf = (unsigned short*)take((size_t)RR * Dd * 2);
  unsigned short* h1_bf   = (unsigned short*)take((size_t)RR * DI * 2);
  (void)ws_size; (void)in_sizes; (void)n_in; (void)out_size;

  const int CB = 256;
  // weights -> bf16, transposed to (N x K) row-major
  cvtT_bf16_kernel<<<(786432 + CB - 1) / CB, CB, 0, stream>>>(Wqkv, wqkvT, Dd, 3 * Hh * DHd);
  for (int h = 0; h < Hh; ++h)
    cvtT_bf16_kernel<<<(4096 + CB - 1) / CB, CB, 0, stream>>>(
        Wmap + h * DHd * Mm, wmapT + h * DHd * Mm, DHd, Mm);
  cvtT_bf16_kernel<<<(262144 + CB - 1) / CB, CB, 0, stream>>>(Wo, woT, Hh * DHd, Dd);
  cvtT_bf16_kernel<<<(1048576 + CB - 1) / CB, CB, 0, stream>>>(W1, w1T, Dd, DI);
  cvtT_bf16_kernel<<<(1048576 + CB - 1) / CB, CB, 0, stream>>>(W2, w2T, DI, Dd);

  // attn_in = center(inp)  (bf16)
  center_bf16_kernel<<<RR, 256, 0, stream>>>(inp, ain_bf, Dd);

  // qkv = attn_in @ Wqkv + bqkv   [8192 x 1536, K=512] -> bf16
  gemm_wmma<true, false, false, false, true><<<dim3(1536 / BN, RR / BM, 1), 128, 0, stream>>>(
      ain_bf, wqkvT, bqkv, nullptr, qkv_bf,
      RR, 1536, Dd, Dd, Dd, 1536, 0, 0, 0);

  // phi_q = elu(q @ Wmap)+1, phi_k = elu(k @ Wmap)+1   (batched over heads)
  gemm_wmma<false, false, true, false, true><<<dim3(1, RR / BM, Hh), 128, 0, stream>>>(
      qkv_bf, wmapT, nullptr, nullptr, phiq_bf,
      RR, Mm, DHd, 3 * Hh * DHd, DHd, Hh * Mm,
      (long)DHd, (long)DHd * Mm, (long)Mm);
  gemm_wmma<false, false, true, false, true><<<dim3(1, RR / BM, Hh), 128, 0, stream>>>(
      qkv_bf + Hh * DHd, wmapT, nullptr, nullptr, phik_bf,
      RR, Mm, DHd, 3 * Hh * DHd, DHd, Hh * Mm,
      (long)DHd, (long)DHd * Mm, (long)Mm);

  // F matrices and V_ext^T
  buildF_kernel<<<(int)(((long)Bb * Hh * Ss * Mm + CB - 1) / CB), CB, 0, stream>>>(
      phiq_bf, phik_bf, pos_ft, slopes, Fq, FkT);
  buildVT_kernel<<<(int)(((long)Bb * Hh * NP * Ss + CB - 1) / CB), CB, 0, stream>>>(
      qkv_bf, VextT);

  // KV^T = Vext^T @ FkT^T   [128 x 256, K=2048] per (b,h) -> bf16
  gemm_wmma<false, false, false, false, true><<<dim3(FEAT / BN, NP / BM, Bb * Hh), 128, 0, stream>>>(
      VextT, FkT, nullptr, nullptr, KVT,
      NP, FEAT, Ss, Ss, Ss, FEAT,
      (long)NP * Ss, (long)FEAT * Ss, (long)NP * FEAT);

  // num = Fq @ KV   [2048 x 128, K=256] per (b,h) -> fp32 (col 64 = z)
  gemm_wmma<false, false, false, false, false><<<dim3(NP / BN, Ss / BM, Bb * Hh), 128, 0, stream>>>(
      Fq, KVT, nullptr, nullptr, numb,
      Ss, NP, FEAT, FEAT, FEAT, NP,
      (long)Ss * FEAT, (long)NP * FEAT, (long)Ss * NP);

  // attn = num / (z + eps)  -> bf16 in (b,s,H*dh) layout
  attn_norm_kernel<<<(int)(((long)Bb * Ss * Hh * DHd + CB - 1) / CB), CB, 0, stream>>>(
      numb, attn_bf);

  // x = attn @ Wo + bo + inp   [8192 x 512, K=512] -> fp32
  gemm_wmma<true, false, false, true, false><<<dim3(Dd / BN, RR / BM, 1), 128, 0, stream>>>(
      attn_bf, woT, bo, inp, xbuf,
      RR, Dd, Hh * DHd, Hh * DHd, Hh * DHd, Dd, 0, 0, 0);

  // ff_in = center(x)  (bf16)
  center_bf16_kernel<<<RR, 256, 0, stream>>>(xbuf, ffin_bf, Dd);

  // h1 = relu(ff_in @ W1 + b1)  [8192 x 2048, K=512] -> bf16
  gemm_wmma<true, true, false, false, true><<<dim3(DI / BN, RR / BM, 1), 128, 0, stream>>>(
      ffin_bf, w1T, b1, nullptr, h1_bf,
      RR, DI, Dd, Dd, Dd, DI, 0, 0, 0);

  // out = h1 @ W2 + b2 + x   [8192 x 512, K=2048] -> fp32
  gemm_wmma<true, false, false, true, false><<<dim3(Dd / BN, RR / BM, 1), 128, 0, stream>>>(
      h1_bf, w2T, b2, xbuf, out,
      RR, Dd, DI, DI, DI, Dd, 0, 0, 0);
}
